// ConstrainedMLP_63797444215547
// MI455X (gfx1250) — matrix operand order repaired and checked by reading, not compiled, not run on hardware
//
#include <hip/hip_runtime.h>
#include <cstdint>

typedef __attribute__((ext_vector_type(2)))  float v2f;
typedef __attribute__((ext_vector_type(4)))  float v4f;
typedef __attribute__((ext_vector_type(8)))  float v8f;
typedef __attribute__((ext_vector_type(16))) __bf16 v16bf;
typedef __attribute__((ext_vector_type(8)))  unsigned short v8us;
typedef __attribute__((ext_vector_type(2)))  unsigned int v2u;

#define EPS_F   0.15f
#define DELTA_F 0.05f
#define N_ITERS 300
#define N_ELEMS 65536
#define NWG     64      // cooperative workgroups in projection
#define PTHREADS 256
#define EPT     4       // elements per thread: 64*256*4 = 65536

// ---------------- workspace layout (bytes) ----------------
static constexpr size_t H1_OFF   = 0;                      // 65536x1024 f32 = 256 MiB
static constexpr size_t H2_OFF   = 268435456;              // 65536x512  f32 = 128 MiB
static constexpr size_t Z0_OFF   = 402653184;              // 65536 f32
static constexpr size_t G8_OFF   = 402915328;              // 65536 u8
static constexpr size_t NS_OFF   = 402980864;              // 16 f32 (n_g, S_g)
static constexpr size_t PART_OFF = 402980928;              // [2][64][8] f32
static constexpr size_t ARR_OFF  = 402985024;              // 300 int
static constexpr size_t DONE_OFF = 402986224;              // 300 int

// split fp32 into bf16 hi (bits[15:0]) + bf16 lo (bits[31:16]) of the result
__device__ __forceinline__ unsigned split_pack(float x) {
    unsigned u = __builtin_bit_cast(unsigned, x);
    float r = x - __builtin_bit_cast(float, u & 0xffff0000u);
    unsigned lu = __builtin_bit_cast(unsigned, r);
    return (u >> 16) | (lu & 0xffff0000u);
}

// pack 4 split results into hi-stream and lo-stream dword pairs
__device__ __forceinline__ void pack4(unsigned s0, unsigned s1, unsigned s2, unsigned s3,
                                      v2u& hv, v2u& lv) {
    unsigned h01 = (s0 & 0xffffu) | (s1 << 16);
    unsigned h23 = (s2 & 0xffffu) | (s3 << 16);
    unsigned l01 = (s0 >> 16)     | (s1 & 0xffff0000u);
    unsigned l23 = (s2 >> 16)     | (s3 & 0xffff0000u);
    hv.x = h01; hv.y = h23;
    lv.x = l01; lv.y = l23;
}

struct us16 { v8us a, b; };
__device__ __forceinline__ v16bf ld_frag(const unsigned short* base, int off0, int off1) {
    us16 t;
    t.a = *(const v8us*)(base + off0);
    t.b = *(const v8us*)(base + off1);
    return __builtin_bit_cast(v16bf, t);
}

// ---------------- GEMM: C = relu(A[MxK]*B[KxN] + bias), split-bf16 WMMA ----------------
// acc += Ah*Bh + Ah*Bl + Al*Bh  with fp32 accumulation (3x v_wmma_f32_16x16x32_bf16)
#define TM 128
#define TN 128
#define TK 32
#define LSTR 56   // ushort stride: 112B rows -> 16B aligned b128, 28*m mod 64 conflict-free

__global__ __launch_bounds__(256)
void gemm_relu_wmma_bf16x3(const float* __restrict__ A, const float* __restrict__ B,
                           const float* __restrict__ bias, float* __restrict__ C,
                           int M, int N, int K, int relu)
{
    __shared__ unsigned short Ahs[TM * LSTR];   // row-major: [m][k]
    __shared__ unsigned short Als[TM * LSTR];
    __shared__ unsigned short Bhs[TN * LSTR];   // col-major: [n][k]
    __shared__ unsigned short Bls[TN * LSTR];

    const int tid  = threadIdx.x;
    const int lane = tid & 31;
    const int wid  = tid >> 5;            // 8 waves
    const int rowTile = blockIdx.y * TM;
    const int colTile = blockIdx.x * TN;

    v8f acc[8] = {};                      // 8 N-subtiles of 16x16 per wave
    const int hi  = lane >> 4;
    const int l15 = lane & 15;
    const int m0  = wid * 16;

    // 16-bit A 16x32 frag: lanes0-15 K 0-15; lanes16-31 K 16-31
    const int aOff = hi * 16;
    // 16-bit B 32x16 frag: lanes0-15 K 0-15; lanes16-31 K 16-31
    const int bOff = hi * 16;

    for (int k0 = 0; k0 < K; k0 += TK) {
        // stage A tile 128x32: 4 float4 per thread, split to bf16 hi/lo, b64 stores
        #pragma unroll
        for (int i = 0; i < 4; ++i) {
            int idx = tid + i * 256;
            int r = idx >> 3;                 // 8 float4 per row
            int c4 = (idx & 7) << 2;
            v4f v = *(const v4f*)&A[(size_t)(rowTile + r) * K + k0 + c4];
            v2u hv, lv;
            pack4(split_pack(v.x), split_pack(v.y), split_pack(v.z), split_pack(v.w), hv, lv);
            *(v2u*)&Ahs[r * LSTR + c4] = hv;
            *(v2u*)&Als[r * LSTR + c4] = lv;
        }
        // stage B tile 32x128 TRANSPOSED into LDS col-major:
        // thread unit = (col, k-quad); global b32 loads coalesced across lanes (consecutive cols)
        #pragma unroll
        for (int i = 0; i < 4; ++i) {
            int idx = tid + i * 256;
            int col = idx & 127;
            int kq  = (idx >> 7) << 2;        // 0,4,...,28
            const float* bp = &B[(size_t)(k0 + kq) * N + colTile + col];
            float f0 = bp[0];
            float f1 = bp[(size_t)N];
            float f2 = bp[(size_t)2 * N];
            float f3 = bp[(size_t)3 * N];
            v2u hv, lv;
            pack4(split_pack(f0), split_pack(f1), split_pack(f2), split_pack(f3), hv, lv);
            *(v2u*)&Bhs[col * LSTR + kq] = hv;
            *(v2u*)&Bls[col * LSTR + kq] = lv;
        }
        __syncthreads();

        // one K=32 span == one bf16 WMMA: 3 passes per subtile
        const int arow = (m0 + l15) * LSTR;
        v16bf ah = ld_frag(Ahs, arow + aOff, arow + aOff + 8);
        v16bf al = ld_frag(Als, arow + aOff, arow + aOff + 8);
        #pragma unroll
        for (int n = 0; n < 8; ++n) {
            const int bcol = (n * 16 + l15) * LSTR;
            v16bf bh = ld_frag(Bhs, bcol + bOff, bcol + bOff + 8);
            v16bf bl = ld_frag(Bls, bcol + bOff, bcol + bOff + 8);
            acc[n] = __builtin_amdgcn_wmma_f32_16x16x32_bf16(
                false, ah, false, bh, (short)0, acc[n], false, false);
            acc[n] = __builtin_amdgcn_wmma_f32_16x16x32_bf16(
                false, ah, false, bl, (short)0, acc[n], false, false);
            acc[n] = __builtin_amdgcn_wmma_f32_16x16x32_bf16(
                false, al, false, bh, (short)0, acc[n], false, false);
        }
        __syncthreads();
    }

    // epilogue: C/D layout — VGPR r: lanes0-15 M=r, lanes16-31 M=r+8
    #pragma unroll
    for (int n = 0; n < 8; ++n) {
        int col = colTile + n * 16 + l15;
        float bv = bias[col];
        #pragma unroll
        for (int r = 0; r < 8; ++r) {
            int row = rowTile + m0 + hi * 8 + r;
            float v = acc[n][r] + bv;
            if (relu) v = fmaxf(v, 0.0f);
            C[(size_t)row * N + col] = v;
        }
    }
}

// ---------------- GEMV + z0 = y/c - 1 : one wave per row ----------------
__global__ __launch_bounds__(256)
void gemv_z0_kernel(const float* __restrict__ H, const float* __restrict__ Wf,
                    const float* __restrict__ bf, const float* __restrict__ c,
                    float* __restrict__ z0, int Kdim)
{
    int wid = threadIdx.x >> 5, lane = threadIdx.x & 31;
    int row = blockIdx.x * 8 + wid;
    const float* hrow = H + (size_t)row * Kdim;
    float s = 0.f;
    for (int t = 0; t < Kdim; t += 128) {
        v4f h = *(const v4f*)&hrow[t + lane * 4];
        v4f w = *(const v4f*)&Wf[t + lane * 4];
        s += h.x * w.x + h.y * w.y + h.z * w.z + h.w * w.w;
    }
    #pragma unroll
    for (int o = 16; o >= 1; o >>= 1) s += __shfl_xor(s, o, 32);
    if (lane == 0) z0[row] = (s + bf[0]) / c[row] - 1.0f;
}

// ---------------- group ids + per-group (n_g, S_g=sum 1/c^2), deterministic ----------------
__global__ __launch_bounds__(1024)
void group_stats_kernel(const float* __restrict__ mask, const float* __restrict__ c,
                        unsigned char* __restrict__ g8, float* __restrict__ nS, int N)
{
    float cnt[8] = {}, S[8] = {};
    for (int n = threadIdx.x; n < N; n += 1024) {
        int g = 0;
        #pragma unroll
        for (int j = 1; j < 8; ++j)
            if (mask[(size_t)j * N + n] > 0.5f) g = j;
        g8[n] = (unsigned char)g;
        float ci = c[n];
        float inv = 1.0f / (ci * ci);
        #pragma unroll
        for (int j = 0; j < 8; ++j) {
            cnt[j] += (g == j) ? 1.0f : 0.0f;
            S[j]   += (g == j) ? inv  : 0.0f;
        }
    }
    #pragma unroll
    for (int o = 16; o >= 1; o >>= 1) {
        #pragma unroll
        for (int j = 0; j < 8; ++j) {
            cnt[j] += __shfl_xor(cnt[j], o, 32);
            S[j]   += __shfl_xor(S[j],   o, 32);
        }
    }
    __shared__ float red[32][16];
    int wid = threadIdx.x >> 5, lane = threadIdx.x & 31;
    if (lane == 0) {
        #pragma unroll
        for (int j = 0; j < 8; ++j) { red[wid][j] = cnt[j]; red[wid][8 + j] = S[j]; }
    }
    __syncthreads();
    if (threadIdx.x < 16) {
        float t = 0.f;
        for (int w = 0; w < 32; ++w) t += red[w][threadIdx.x];
        nS[threadIdx.x] = t;
    }
}

// ---------------- persistent cooperative Dykstra projection ----------------
__global__ __launch_bounds__(PTHREADS)
void dykstra_kernel(const float* __restrict__ z0, const float* __restrict__ c,
                    const unsigned char* __restrict__ g8, const float* __restrict__ nS,
                    float* __restrict__ partials, int* __restrict__ arrive,
                    int* __restrict__ done, float* __restrict__ out)
{
    const int tid  = threadIdx.x;
    const int wg   = blockIdx.x;
    const int gtid = wg * PTHREADS + tid;     // 0..16383
    const int lane = tid & 31, wid = tid >> 5;

    __shared__ float eBuf[2][8];
    __shared__ float warpAcc[8][8];
    __shared__ float nsh[16];
    if (tid < 16) nsh[tid] = nS[tid];
    if (tid < 8)  { eBuf[0][tid] = 0.f; eBuf[1][tid] = 0.f; }
    __syncthreads();

    float z[EPT], p[EPT], ug[EPT], cv[EPT];
    int   g[EPT];
    #pragma unroll
    for (int j = 0; j < EPT; ++j) {
        int e = gtid + j * (NWG * PTHREADS);
        z[j] = z0[e];
        p[j] = 0.f;
        g[j] = (int)g8[e];
        cv[j] = c[e];
        ug[j] = nsh[g[j]] / (cv[j] * cv[j] * nsh[8 + g[j]]);   // n_g / (c^2 S_g)
    }

    for (int it = 0; it < N_ITERS; ++it) {
        const int par = it & 1;
        if (it >= 2) {                        // guard partial-buffer reuse (distance-2)
            if (tid == 0) {
                while (__hip_atomic_load(&done[it - 2], __ATOMIC_ACQUIRE,
                                         __HIP_MEMORY_SCOPE_AGENT) < NWG)
                    __builtin_amdgcn_s_sleep(1);
            }
            __syncthreads();
        }
        // pass 1: box projection + group partial sums of t = zb + u*e_prev
        float acc[8] = {};
        #pragma unroll
        for (int j = 0; j < EPT; ++j) {
            float s  = z[j] + p[j];
            float zb = fminf(fmaxf(s, -EPS_F), EPS_F);
            p[j] = s - zb;
            z[j] = zb;
            float t = zb + ug[j] * eBuf[par][g[j]];
            #pragma unroll
            for (int q = 0; q < 8; ++q) acc[q] += (g[j] == q) ? t : 0.0f;
        }
        #pragma unroll
        for (int o = 16; o >= 1; o >>= 1) {
            #pragma unroll
            for (int q = 0; q < 8; ++q) acc[q] += __shfl_xor(acc[q], o, 32);
        }
        if (lane == 0) {
            #pragma unroll
            for (int q = 0; q < 8; ++q) warpAcc[wid][q] = acc[q];
        }
        __syncthreads();
        if (wid == 0 && lane < 8) {
            float t = 0.f;
            #pragma unroll
            for (int w = 0; w < 8; ++w) t += warpAcc[w][lane];
            partials[(size_t)par * NWG * 8 + wg * 8 + lane] = t;
        }
        if (tid == 0) {
            __hip_atomic_fetch_add(&arrive[it], 1, __ATOMIC_RELEASE,
                                   __HIP_MEMORY_SCOPE_AGENT);
            while (__hip_atomic_load(&arrive[it], __ATOMIC_ACQUIRE,
                                     __HIP_MEMORY_SCOPE_AGENT) < NWG)
                __builtin_amdgcn_s_sleep(1);
        }
        __syncthreads();
        // every WG redundantly reduces 64x8 partials in identical fixed order
        if (wid == 0 && lane < 8) {
            float tot = 0.f;
            for (int w = 0; w < NWG; ++w)
                tot += __hip_atomic_load(&partials[(size_t)par * NWG * 8 + w * 8 + lane],
                                         __ATOMIC_RELAXED, __HIP_MEMORY_SCOPE_AGENT);
            float v  = tot / nsh[lane];
            float cl = fminf(fmaxf(v, -DELTA_F), DELTA_F);
            eBuf[par ^ 1][lane] = v - cl;
        }
        __syncthreads();
        if (tid == 0)
            __hip_atomic_fetch_add(&done[it], 1, __ATOMIC_RELEASE,
                                   __HIP_MEMORY_SCOPE_AGENT);
        // pass 2: zg = t - u*e_new = zb + u*(e_prev - e_new)
        #pragma unroll
        for (int j = 0; j < EPT; ++j)
            z[j] += ug[j] * (eBuf[par][g[j]] - eBuf[par ^ 1][g[j]]);
    }

    #pragma unroll
    for (int j = 0; j < EPT; ++j) {
        int e = gtid + j * (NWG * PTHREADS);
        out[e] = (z[j] + 1.0f) * cv[j];
    }
}

// ---------------- host launcher ----------------
extern "C" void kernel_launch(void* const* d_in, const int* in_sizes, int n_in,
                              void* d_out, int out_size, void* d_ws, size_t ws_size,
                              hipStream_t stream) {
    const float* x    = (const float*)d_in[0];
    const float* W1   = (const float*)d_in[1];
    const float* b1   = (const float*)d_in[2];
    const float* W2   = (const float*)d_in[3];
    const float* b2   = (const float*)d_in[4];
    const float* Wf   = (const float*)d_in[5];
    const float* bf   = (const float*)d_in[6];
    const float* c    = (const float*)d_in[7];
    const float* mask = (const float*)d_in[8];

    char* ws = (char*)d_ws;
    float*         h1   = (float*)(ws + H1_OFF);
    float*         h2   = (float*)(ws + H2_OFF);
    float*         z0   = (float*)(ws + Z0_OFF);
    unsigned char* g8   = (unsigned char*)(ws + G8_OFF);
    float*         nS   = (float*)(ws + NS_OFF);
    float*         part = (float*)(ws + PART_OFF);
    int*           arr  = (int*)(ws + ARR_OFF);
    int*           don  = (int*)(ws + DONE_OFF);
    float*         out  = (float*)d_out;

    // reset cooperative counters every launch (graph-capturable)
    (void)hipMemsetAsync(ws + ARR_OFF, 0, 2 * N_ITERS * sizeof(int), stream);

    // layer 1: h1 = relu(x @ W1 + b1)   [65536x512]x[512x1024]
    gemm_relu_wmma_bf16x3<<<dim3(1024 / TN, N_ELEMS / TM), 256, 0, stream>>>(
        x, W1, b1, h1, N_ELEMS, 1024, 512, 1);
    // layer 2: h2 = relu(h1 @ W2 + b2)  [65536x1024]x[1024x512]
    gemm_relu_wmma_bf16x3<<<dim3(512 / TN, N_ELEMS / TM), 256, 0, stream>>>(
        h1, W2, b2, h2, N_ELEMS, 512, 1024, 1);
    // final: z0 = (h2 @ Wf + bf)/c - 1
    gemv_z0_kernel<<<N_ELEMS / 8, 256, 0, stream>>>(h2, Wf, bf, c, z0, 512);
    // group ids + deterministic group stats
    group_stats_kernel<<<1, 1024, 0, stream>>>(mask, c, g8, nS, N_ELEMS);
    // 300-iteration Dykstra projection, cooperative 64-WG kernel
    dykstra_kernel<<<NWG, PTHREADS, 0, stream>>>(z0, c, g8, nS, part, arr, don, out);
}
